// AttentionLayer_22952305230312
// MI455X (gfx1250) — matrix operand order repaired and checked by reading, not compiled, not run on hardware
//
#include <hip/hip_runtime.h>
#include <hip/hip_bf16.h>

// ---------------------------------------------------------------------------
// MI455X (gfx1250) attention layer, bf16 WMMA pipeline, wave32.
//   K0: convert x -> bf16; build transposed bf16 weight tables
//   K1: QKV projection GEMM (v_wmma_f32_16x16x32_bf16), writes q,k,[v^T] bf16
//   K2: attention: block=(head, s-tile), wave=batch; exact batch-axis softmax
//       via LDS exchange; scores + P*V on WMMA; writes heads_cat bf16
//   K3: output projection GEMM (WMMA) with async-to-LDS A-panel staging
//       + bias + LayerNorm + LeakyReLU -> f32
// ---------------------------------------------------------------------------

#define B_   16
#define S_   1024
#define IN_  512
#define OUT_ 512
#define H_   8
#define DH_  64
#define BS_  (B_ * S_)

typedef __bf16 v16bf __attribute__((ext_vector_type(16)));
typedef __bf16 v8bf  __attribute__((ext_vector_type(8)));
typedef __bf16 v2bf  __attribute__((ext_vector_type(2)));
typedef float  v8f   __attribute__((ext_vector_type(8)));

static __device__ __forceinline__ v8f wmma_bf16(v16bf a, v16bf b, v8f c) {
  // D(16x16,f32) = A(16x32,bf16) * B(32x16,bf16) + C
  return __builtin_amdgcn_wmma_f32_16x16x32_bf16(false, a, false, b,
                                                 (short)0, c, false, false);
}

// A-matrix element -> K index (16-bit A, 16x32; this file's contiguous order k = 16*half + e, B loads match)
static __device__ __forceinline__ int kA(int e, int half) {
  return e + half * 16;
}

// Load contiguous 16 bf16 (16B-aligned) into a v16bf (B-matrix tile row run)
static __device__ __forceinline__ v16bf load16(const __bf16* p) {
  const v8bf* vp = (const v8bf*)p;
  v8bf lo = vp[0], hi = vp[1];
  v16bf r;
#pragma unroll
  for (int e = 0; e < 8; ++e) { r[e] = lo[e]; r[8 + e] = hi[e]; }
  return r;
}

// Load A-tile from a row-major row (stride-1 in K), pairwise (4B) loads
static __device__ __forceinline__ v16bf loadA(const __bf16* row, int half) {
  v16bf a;
#pragma unroll
  for (int e = 0; e < 16; e += 2) {
    v2bf p = *(const v2bf*)(row + kA(e, half));
    a[e] = p[0]; a[e + 1] = p[1];
  }
  return a;
}

// ---------------------------------------------------------------------------
// K0a: x (f32) -> xb (bf16)
__global__ void conv_x_kernel(const float* __restrict__ x, __bf16* __restrict__ xb) {
  int i = blockIdx.x * blockDim.x + threadIdx.x;
  if (i < BS_ * IN_) xb[i] = (__bf16)x[i];
}

// K0b: build wt[j][f] = w_{proj}[h][f][d] (j = proj*512 + h*64 + d), bf16,
//      and wobT[o][f] = w_out[f][o], bf16. Both transposed so B-tiles are
//      contiguous in K.
__global__ void conv_w_kernel(const float* __restrict__ wq, const float* __restrict__ wk,
                              const float* __restrict__ wv, const float* __restrict__ w_out,
                              __bf16* __restrict__ wt, __bf16* __restrict__ wobT) {
  int i = blockIdx.x * blockDim.x + threadIdx.x;
  const int NW = 3 * OUT_ * IN_;              // 1536*512
  if (i < NW) {
    int j = i >> 9, f = i & 511;
    int proj = j >> 9, h = (j >> 6) & 7, d = j & 63;
    const float* src = (proj == 0) ? wq : (proj == 1) ? wk : wv;
    wt[(size_t)j * IN_ + f] = (__bf16)src[((size_t)h * IN_ + f) * DH_ + d];
  } else if (i < NW + OUT_ * OUT_) {
    int t = i - NW;
    int o = t >> 9, f = t & 511;
    wobT[(size_t)o * OUT_ + f] = (__bf16)w_out[(size_t)f * OUT_ + o];
  }
}

// ---------------------------------------------------------------------------
// K1: QKV GEMM: [16384,512] x [512,1536] in 16x16 tiles, one tile per wave.
//     q,k stored [h][b][s][d] bf16; v stored transposed [h][b][d][s] bf16.
__global__ void qkv_gemm_kernel(const __bf16* __restrict__ xb, const __bf16* __restrict__ wt,
                                __bf16* __restrict__ q, __bf16* __restrict__ k,
                                __bf16* __restrict__ vT) {
  const int tid  = threadIdx.x;
  const int wave = tid >> 5;
  const int lane = tid & 31;
  const int n    = lane & 15;
  const int half = lane >> 4;

  const int row0 = blockIdx.x * 16;                 // flat (b,s) row tile
  const int ct   = blockIdx.y * 8 + wave;           // column tile 0..95
  const int c0   = ct * 16;

  v8f acc = {};
#pragma unroll 4
  for (int kb = 0; kb < IN_ / 32; ++kb) {
    const int k0 = kb * 32;
    const __bf16* arow = xb + (size_t)(row0 + n) * IN_ + k0;
    v16bf a = loadA(arow, half);
    const __bf16* brow = wt + (size_t)(c0 + n) * IN_ + k0 + half * 16;
    __builtin_prefetch(brow + 32, 0, 1);            // global_prefetch_b8
    v16bf b = load16(brow);
    acc = wmma_bf16(a, b, acc);
  }

  const int proj = c0 >> 9;
  const int h    = (c0 >> 6) & 7;
  const int d0   = c0 & 63;
  const int bq   = row0 >> 10;
  const int s0   = row0 & 1023;

#pragma unroll
  for (int r = 0; r < 8; ++r) {
    const int s = s0 + r + 8 * half;
    const __bf16 bv = (__bf16)acc[r];
    if (proj == 0)
      q[(((size_t)(h * B_ + bq)) * S_ + s) * DH_ + d0 + n] = bv;
    else if (proj == 1)
      k[(((size_t)(h * B_ + bq)) * S_ + s) * DH_ + d0 + n] = bv;
    else
      vT[(((size_t)(h * B_ + bq)) * DH_ + d0 + n) * S_ + s] = bv;
  }
}

// ---------------------------------------------------------------------------
// K2: attention. grid=(S/16, H); block = 512 threads = 16 waves; wave = batch.
//     Softmax is over the BATCH axis (faithful to reference) -> fully local
//     per (s,t): exchange 16x32 score tiles via LDS, exact softmax, resume.
__global__ void attn_kernel(const __bf16* __restrict__ q, const __bf16* __restrict__ k,
                            const __bf16* __restrict__ vT, __bf16* __restrict__ hcb) {
  __shared__ float  sred[B_][16][32];   // 32 KB: raw scores, all batches
  __shared__ __bf16 pbuf[B_][16][32];   // 16 KB: probabilities, all batches

  const int tid  = threadIdx.x;
  const int bw   = tid >> 5;            // wave index == batch
  const int lane = tid & 31;
  const int n    = lane & 15;
  const int half = lane >> 4;

  const int s0 = blockIdx.x * 16;
  const int h  = blockIdx.y;

  const size_t hb = (size_t)(h * B_ + bw);

  // Q A-tiles for this wave's batch: K-dim = d, split 0..31 / 32..63
  const __bf16* qrow = q + (hb * S_ + s0 + n) * DH_;
  v16bf aq0 = loadA(qrow, half);
  v16bf aq1 = loadA(qrow + 32, half);

  v8f o0 = {}, o1 = {}, o2 = {}, o3 = {};

  for (int t0 = 0; t0 < S_; t0 += 32) {
    // --- scores for two 16-wide t tiles ---
#pragma unroll
    for (int tt = 0; tt < 2; ++tt) {
      const int tc0 = t0 + tt * 16;
      const __bf16* krow = k + (hb * S_ + tc0 + n) * DH_ + half * 16;
      v16bf bk0 = load16(krow);
      v16bf bk1 = load16(krow + 32);
      v8f sc = {};
      sc = wmma_bf16(aq0, bk0, sc);
      sc = wmma_bf16(aq1, bk1, sc);
#pragma unroll
      for (int r = 0; r < 8; ++r)
        sred[bw][r + 8 * half][tt * 16 + n] = sc[r];
    }
    __syncthreads();

    // --- exact softmax over the 16 batches; one element per thread ---
    {
      const int mrow = tid >> 5;        // 0..15
      const int tc   = tid & 31;        // 0..31
      float mx = -3.0e38f;
      float vb[B_];
#pragma unroll
      for (int bb = 0; bb < B_; ++bb) {
        vb[bb] = sred[bb][mrow][tc] * 0.125f;   // 1/sqrt(DH)
        mx = fmaxf(mx, vb[bb]);
      }
      float sum = 0.f;
#pragma unroll
      for (int bb = 0; bb < B_; ++bb) { vb[bb] = __expf(vb[bb] - mx); sum += vb[bb]; }
      const float inv = 1.0f / sum;
#pragma unroll
      for (int bb = 0; bb < B_; ++bb)
        pbuf[bb][mrow][tc] = (__bf16)(vb[bb] * inv);
    }
    __syncthreads();

    // --- P(16x32) * V(32x64): A from LDS, B from vT (contiguous in t) ---
    v16bf ap;
#pragma unroll
    for (int e = 0; e < 16; ++e) ap[e] = pbuf[bw][n][kA(e, half)];

#pragma unroll
    for (int c = 0; c < 4; ++c) {
      const __bf16* vrow = vT + (hb * DH_ + c * 16 + n) * S_ + t0 + half * 16;
      v16bf bv = load16(vrow);
      v8f* oc = (c == 0) ? &o0 : (c == 1) ? &o1 : (c == 2) ? &o2 : &o3;
      *oc = wmma_bf16(ap, bv, *oc);
    }
    // next-iter barrier 1 protects pbuf reads; sred(i+1) writes are safe (post barrier2)
  }

  // heads_cat[b][s][h*64+d], bf16
#pragma unroll
  for (int c = 0; c < 4; ++c) {
    const v8f oc = (c == 0) ? o0 : (c == 1) ? o1 : (c == 2) ? o2 : o3;
#pragma unroll
    for (int r = 0; r < 8; ++r) {
      const int s = s0 + r + 8 * half;
      const int d = h * DH_ + c * 16 + n;
      hcb[((size_t)bw * S_ + s) * OUT_ + d] = (__bf16)oc[r];
    }
  }
}

// ---------------------------------------------------------------------------
// K3: y = heads_cat @ w_out + b_out; LayerNorm; LeakyReLU(0.1) -> f32 out.
//     block = 512 threads = 16 waves; each wave: 2 column tiles of the GEMM.
//     The 16x512 bf16 A-panel is staged into LDS ONCE via async-to-LDS DMA
//     (global_load_async_to_lds_b128, ASYNCcnt) so 16 waves don't re-read it
//     16x from global. Then wave w performs the LN reduction for row w.
__global__ void out_ln_kernel(const __bf16* __restrict__ hcb, const __bf16* __restrict__ wobT,
                              const float* __restrict__ b_out, const float* __restrict__ gamma,
                              const float* __restrict__ beta, float* __restrict__ out) {
  __shared__ __bf16 apan[16][OUT_];     // 16 KB: staged A panel (16 rows x 512)
  __shared__ float  ybuf[16][OUT_];     // 32 KB

  const int tid  = threadIdx.x;
  const int wave = tid >> 5;
  const int lane = tid & 31;
  const int n    = lane & 15;
  const int half = lane >> 4;

  const int row0 = blockIdx.x * 16;

  // --- async stage of A panel: 16 KB = 1024 x 16B chunks, 2 per thread ---
#pragma unroll
  for (int j = 0; j < 2; ++j) {
    const int c   = tid + j * 512;      // chunk id 0..1023
    const int r   = c >> 6;             // row 0..15 (64 chunks per 1KB row)
    const int cc  = (c & 63) * 8;       // element offset within row
    const __bf16* gsrc = hcb + (size_t)(row0 + r) * OUT_ + cc;
    unsigned ldst = (unsigned)(size_t)&apan[r][cc];   // LDS byte offset
    asm volatile("global_load_async_to_lds_b128 %0, %1, off"
                 :: "v"(ldst), "v"(gsrc) : "memory");
  }
  asm volatile("s_wait_asynccnt 0x0" ::: "memory");
  __syncthreads();

  v8f acc0 = {}, acc1 = {};
#pragma unroll 4
  for (int kb = 0; kb < OUT_ / 32; ++kb) {
    const int k0 = kb * 32;
    v16bf a = loadA(&apan[n][k0], half);            // ds_load from staged panel
    const __bf16* brow0 = wobT + (size_t)(wave * 32 + n) * OUT_ + k0 + half * 16;
    const __bf16* brow1 = wobT + (size_t)(wave * 32 + 16 + n) * OUT_ + k0 + half * 16;
    __builtin_prefetch(brow0 + 32, 0, 1);
    v16bf b0 = load16(brow0);
    v16bf b1 = load16(brow1);
    acc0 = wmma_bf16(a, b0, acc0);
    acc1 = wmma_bf16(a, b1, acc1);
  }

#pragma unroll
  for (int j = 0; j < 2; ++j) {
    const int c0 = wave * 32 + j * 16;
    const v8f av = (j == 0) ? acc0 : acc1;
    const float bias = b_out[c0 + n];
#pragma unroll
    for (int r = 0; r < 8; ++r)
      ybuf[r + 8 * half][c0 + n] = av[r] + bias;
  }
  __syncthreads();

  // LayerNorm: wave w owns row w; 32 lanes x 16 elems each
  float sum = 0.f, sq = 0.f;
#pragma unroll
  for (int i = 0; i < 16; ++i) {
    const float v = ybuf[wave][lane + i * 32];
    sum += v; sq += v * v;
  }
#pragma unroll
  for (int off = 16; off >= 1; off >>= 1) {
    sum += __shfl_xor(sum, off, 32);
    sq  += __shfl_xor(sq,  off, 32);
  }
  const float mean = sum * (1.0f / OUT_);
  const float var  = sq * (1.0f / OUT_) - mean * mean;
  const float rstd = rsqrtf(var + 1e-5f);

#pragma unroll
  for (int i = 0; i < 16; ++i) {
    const int c = lane + i * 32;
    float v = (ybuf[wave][c] - mean) * rstd * gamma[c] + beta[c];
    v = (v >= 0.f) ? v : 0.1f * v;
    out[(size_t)(row0 + wave) * OUT_ + c] = v;
  }
}

// ---------------------------------------------------------------------------
extern "C" void kernel_launch(void* const* d_in, const int* in_sizes, int n_in,
                              void* d_out, int out_size, void* d_ws, size_t ws_size,
                              hipStream_t stream) {
  const float* x      = (const float*)d_in[0];
  const float* wq     = (const float*)d_in[1];
  const float* wk     = (const float*)d_in[2];
  const float* wv     = (const float*)d_in[3];
  const float* w_out  = (const float*)d_in[4];
  const float* b_out  = (const float*)d_in[5];
  const float* gamma  = (const float*)d_in[6];
  const float* beta   = (const float*)d_in[7];
  float* out = (float*)d_out;

  char* ws = (char*)d_ws;
  size_t off = 0;
  __bf16* xb   = (__bf16*)(ws + off); off += (size_t)BS_ * IN_ * 2;          // 16.8 MB
  __bf16* wt   = (__bf16*)(ws + off); off += (size_t)3 * OUT_ * IN_ * 2;     //  1.6 MB
  __bf16* wobT = (__bf16*)(ws + off); off += (size_t)OUT_ * OUT_ * 2;        //  0.5 MB
  __bf16* q    = (__bf16*)(ws + off); off += (size_t)H_ * B_ * S_ * DH_ * 2; // 16.8 MB
  __bf16* k    = (__bf16*)(ws + off); off += (size_t)H_ * B_ * S_ * DH_ * 2; // 16.8 MB
  __bf16* vT   = (__bf16*)(ws + off); off += (size_t)H_ * B_ * S_ * DH_ * 2; // 16.8 MB
  __bf16* hcb  = (__bf16*)(ws + off); off += (size_t)BS_ * OUT_ * 2;         // 16.8 MB

  // K0: conversions
  conv_x_kernel<<<(BS_ * IN_) / 256, 256, 0, stream>>>(x, xb);
  conv_w_kernel<<<(3 * OUT_ * IN_ + OUT_ * OUT_) / 256, 256, 0, stream>>>(
      wq, wk, wv, w_out, wt, wobT);

  // K1: QKV projections (1024 row tiles x 96 col tiles, 8 waves/block)
  qkv_gemm_kernel<<<dim3(BS_ / 16, 12), 256, 0, stream>>>(xb, wt, q, k, vT);

  // K2: attention with batch-axis softmax (block per (s-tile, head))
  attn_kernel<<<dim3(S_ / 16, H_), 512, 0, stream>>>(q, k, vT, hcb);

  // K3: output projection + LayerNorm + LeakyReLU
  out_ln_kernel<<<BS_ / 16, 512, 0, stream>>>(hcb, wobT, b_out, gamma, beta, out);
}
